// CausalSelfAttention_5214090298017
// MI455X (gfx1250) — matrix operand order repaired and checked by reading, not compiled, not run on hardware
//
#include <hip/hip_runtime.h>
#include <stdint.h>

// Problem constants (fixed by the reference)
#define Bb 2
#define Tt 2048
#define Cc 768
#define Hh 12
#define HDd 64
constexpr int GM = Bb * Tt;   // 4096 rows for all GEMMs
constexpr int GN = Cc;        // 768
constexpr int GK = Cc;        // 768

typedef unsigned short ushort_t;
typedef __attribute__((ext_vector_type(16))) __bf16 v16bf;
typedef __attribute__((ext_vector_type(8)))  float  v8f;
typedef __attribute__((ext_vector_type(4)))  unsigned int v4u;
typedef __attribute__((ext_vector_type(8)))  int v8i;
typedef __attribute__((ext_vector_type(4)))  int v4i;

// 16x32 (A) / 32x16 (B) bf16 fragment: 8 VGPRs
union Frag16 {
    v16bf    v;
    ushort_t us[16];
    uint4    q[2];
};
// 16x16 f32 C/D fragment: 8 VGPRs
union FragF {
    v8f   v;
    float f[8];
};

__device__ __forceinline__ ushort_t f2bf(float x) {
    union { float f; uint32_t u; } c; c.f = x;
    uint32_t u = c.u;
    return (ushort_t)((u + 0x7FFFu + ((u >> 16) & 1u)) >> 16);  // RNE
}

// LDS byte offset of a generic pointer to __shared__ memory.
// ISA 10.2: for LDS-aperture addresses, the low 32 address bits are the LDS offset.
__device__ __forceinline__ uint32_t lds_byte_addr(const void* p) {
    return (uint32_t)(uintptr_t)p;
}

// ---------------------------------------------------------------------------
// TDM issue: 2B elements, one descriptor; group1 words 1..5 parameterized.
// g1 layout (ISA 8.4): w0={pad/flags,data_size,wg_mask}, w1[31:16]=dim0 lo,
// w2={dim1 lo16<<16 | dim0 hi16}, w3={tile_dim0<<16 | dim1 hi16},
// w4={tile_dim2<<16 | tile_dim1}, w5=stride0 lo32.
// ---------------------------------------------------------------------------
__device__ __forceinline__ void tdm_issue(uint32_t lds_byte, const void* gptr,
                                          int w1, int w2, int w3, int w4,
                                          int w5) {
    uint64_t ga = (uint64_t)(uintptr_t)gptr;
    v4u g0;
    g0[0] = 1u;                                              // count=1
    g0[1] = lds_byte;                                        // lds_addr
    g0[2] = (uint32_t)ga;                                    // global_addr lo
    g0[3] = ((uint32_t)(ga >> 32) & 0x01FFFFFFu) | 0x80000000u;  // hi + type=2
    v8i g1;
    g1[0] = 0x00010000;  // data_size=1 (2 bytes), wg_mask=0, no pad/iterate
    g1[1] = w1; g1[2] = w2; g1[3] = w3; g1[4] = w4; g1[5] = w5;
    g1[6] = 0;  g1[7] = 0;
    v4i z4 = (v4i){0, 0, 0, 0};
#if defined(__clang_major__) && (__clang_major__ >= 23)
    v8i z8 = (v8i){0, 0, 0, 0, 0, 0, 0, 0};
    __builtin_amdgcn_tensor_load_to_lds(g0, g1, z4, z4, z8, 0);
#else
    __builtin_amdgcn_tensor_load_to_lds(g0, g1, z4, z4, 0);
#endif
}

// K tile: contiguous 32 rows x 64 d = 2048 elements (1-D copy)
__device__ __forceinline__ void tdm_issue_k(uint32_t lds_byte,
                                            const ushort_t* g) {
    tdm_issue(lds_byte, g, (int)(2048u << 16), 0, (int)(2048u << 16), 0, 2048);
}
// V tile from Vt: 64 rows(d) x 32 cols(t), row stride T=2048 elements
__device__ __forceinline__ void tdm_issue_v(uint32_t lds_byte,
                                            const ushort_t* g) {
    tdm_issue(lds_byte, g, (int)(32u << 16), (int)(64u << 16),
              (int)(32u << 16), 64, 2048);
}

// ---------------------------------------------------------------------------
// Phase 0: f32 -> bf16 bulk convert (x and the four weight matrices)
// ---------------------------------------------------------------------------
__global__ void f32_to_bf16_kernel(const float* __restrict__ in,
                                   ushort_t* __restrict__ out, int n) {
    int i = blockIdx.x * blockDim.x + threadIdx.x;
    if (i < n) out[i] = f2bf(in[i]);
}

// ---------------------------------------------------------------------------
// Phase 1/3: OUT = A[M,K] * W[N,K]^T + bias   via v_wmma_f32_16x16x32_bf16
//   MODE_QK : store bf16 to [B,H,T,HD]
//   MODE_VT : store bf16 to [B,H,HD,T]  (V transposed for the PV matmul)
//   MODE_F32: store f32 row-major [M,N] (final projection -> d_out)
// Block: 256 threads = 8 waves; tile 256(M) x 64(N); wave = 32(M) x 64(N),
// i.e. 8 WMMAs per K-step sharing one A-pair and four B fragments.
// ---------------------------------------------------------------------------
enum { MODE_QK = 0, MODE_VT = 1, MODE_F32 = 2 };

template <int MODE>
__global__ __launch_bounds__(256) void gemm_wmma_kernel(
    const ushort_t* __restrict__ A, const ushort_t* __restrict__ W,
    const float* __restrict__ bias, void* __restrict__ outp) {
    const int lane = threadIdx.x & 31;
    const int wave = threadIdx.x >> 5;
    const int lr = lane & 15;   // A: row in tile / B: col in tile
    const int lh = lane >> 4;   // selects K-half per the 16-bit layouts
    const int m0 = blockIdx.x * 256 + wave * 32;
    const int n0 = blockIdx.y * 64;

    FragF acc[2][4];
#pragma unroll
    for (int i = 0; i < 2; ++i)
#pragma unroll
        for (int j = 0; j < 4; ++j)
#pragma unroll
            for (int r = 0; r < 8; ++r) acc[i][j].f[r] = 0.0f;

    // A fragments: rows m0+lr and m0+16+lr; K runs at k0+lh*8 / k0+16+lh*8
    const ushort_t* arow0 = A + (size_t)(m0 + lr) * GK + lh * 8;
    const ushort_t* arow1 = arow0 + (size_t)16 * GK;

    for (int k0 = 0; k0 < GK; k0 += 32) {
        __builtin_prefetch(arow0 + k0 + 64, 0, 3);  // global_prefetch_b8
        Frag16 a0, a1;
        a0.q[0] = *(const uint4*)(arow0 + k0);
        a0.q[1] = *(const uint4*)(arow0 + k0 + 16);
        a1.q[0] = *(const uint4*)(arow1 + k0);
        a1.q[1] = *(const uint4*)(arow1 + k0 + 16);
#pragma unroll
        for (int j = 0; j < 4; ++j) {
            const ushort_t* wrow =
                W + (size_t)(n0 + j * 16 + lr) * GK + k0 + lh * 8;
            Frag16 bfr;
            bfr.q[0] = *(const uint4*)(wrow);
            bfr.q[1] = *(const uint4*)(wrow + 16);
            acc[0][j].v = __builtin_amdgcn_wmma_f32_16x16x32_bf16(
                false, a0.v, false, bfr.v, (short)0, acc[0][j].v, false, false);
            acc[1][j].v = __builtin_amdgcn_wmma_f32_16x16x32_bf16(
                false, a1.v, false, bfr.v, (short)0, acc[1][j].v, false, false);
        }
    }

#pragma unroll
    for (int j = 0; j < 4; ++j) {
        const int n = n0 + j * 16 + lr;
        const float bv = bias[n];
#pragma unroll
        for (int i = 0; i < 2; ++i)
#pragma unroll
            for (int r = 0; r < 8; ++r) {
                const int m = m0 + i * 16 + r + (lh << 3);
                const float val = acc[i][j].f[r] + bv;
                if (MODE == MODE_F32) {
                    ((float*)outp)[(size_t)m * GN + n] = val;
                } else {
                    const int b = m >> 11;  // m / T
                    const int t = m & (Tt - 1);
                    const int h = n / HDd;
                    const int hd = n & (HDd - 1);
                    size_t idx;
                    if (MODE == MODE_QK)
                        idx = (((size_t)b * Hh + h) * Tt + t) * HDd + hd;
                    else
                        idx = (((size_t)b * Hh + h) * HDd + hd) * Tt + t;
                    ((ushort_t*)outp)[idx] = f2bf(val);
                }
            }
    }
}

// ---------------------------------------------------------------------------
// Phase 2: flash attention, TDM-fed. One wave per 16-query tile per (b,h).
// Q,K: [B,H,T,HD] bf16.  Vt: [B,H,HD,T] bf16.  mask: [B,T] i32 (0 => -inf).
// K tiles double-buffered via tensor_load_to_lds; V tile single-buffered.
// ---------------------------------------------------------------------------
__global__ __launch_bounds__(128) void attn_kernel(
    const ushort_t* __restrict__ Q, const ushort_t* __restrict__ K,
    const ushort_t* __restrict__ Vt, const int* __restrict__ mask,
    ushort_t* __restrict__ Y) {
    __shared__ ushort_t lds_k[4][2][32 * 64];  // 32 KB: K tiles, double-buffer
    __shared__ ushort_t lds_v[4][64 * 32];     // 16 KB: V tile
    __shared__ ushort_t lds_pt[4][16 * 32];    // 4 KB : P re-layout

    const int lane = threadIdx.x & 31;
    const int wave = threadIdx.x >> 5;
    const int lr = lane & 15;
    const int lh = lane >> 4;

    const int tile = blockIdx.x * 4 + wave;   // 0 .. B*H*T/16-1
    const int qt = tile & (Tt / 16 - 1);      // query tile in head
    const int bh = tile >> 7;                 // b*H + h
    const int b = bh / Hh;
    const int h = bh - b * Hh;

    const ushort_t* Qb = Q + (size_t)bh * Tt * HDd;
    const ushort_t* Kb = K + (size_t)bh * Tt * HDd;
    const ushort_t* Vb = Vt + (size_t)bh * HDd * Tt;
    const int* mb = mask + b * Tt;

    // Q A-fragments for d=[0,32) and d=[32,64)
    Frag16 aq0, aq1;
    {
        const ushort_t* qrow = Qb + (size_t)(qt * 16 + lr) * HDd + lh * 16;
        aq0.q[0] = *(const uint4*)(qrow);
        aq0.q[1] = *(const uint4*)(qrow + 8);
        aq1.q[0] = *(const uint4*)(qrow + 32);
        aq1.q[1] = *(const uint4*)(qrow + 40);
    }

    FragF acc[4];
#pragma unroll
    for (int j = 0; j < 4; ++j)
#pragma unroll
        for (int r = 0; r < 8; ++r) acc[j].f[r] = 0.0f;

    float run_max[8], run_sum[8];
#pragma unroll
    for (int r = 0; r < 8; ++r) { run_max[r] = -1e30f; run_sum[r] = 0.0f; }

    const float scale = 0.125f;  // 1/sqrt(64)
    ushort_t* pl = lds_pt[wave];
    const uint32_t ldsK0 = lds_byte_addr(&lds_k[wave][0][0]);
    const uint32_t ldsK1 = lds_byte_addr(&lds_k[wave][1][0]);
    const uint32_t ldsV  = lds_byte_addr(&lds_v[wave][0]);

    // Prologue: stage first K tile
    tdm_issue_k(ldsK0, Kb);

    for (int kb = 0; kb < Tt; kb += 32) {
        const int ib = (kb >> 5) & 1;
        // Stage current V tile; prefetch next K tile; wait for current pair.
        tdm_issue_v(ldsV, Vb + kb);
        if (kb + 32 < Tt) {
            tdm_issue_k(ib ? ldsK0 : ldsK1, Kb + (size_t)(kb + 32) * HDd);
            __builtin_amdgcn_s_wait_tensorcnt(1);  // current K+V landed
        } else {
            __builtin_amdgcn_s_wait_tensorcnt(0);
        }

        const ushort_t* kt = &lds_k[wave][ib][0];
        const ushort_t* vt = &lds_v[wave][0];

        // --- S = Q K^T for 32 keys (two 16-key column groups), from LDS ---
        FragF s0, s1;
#pragma unroll
        for (int r = 0; r < 8; ++r) { s0.f[r] = 0.0f; s1.f[r] = 0.0f; }
        {
            const ushort_t* kr0 = kt + lr * 64 + lh * 16;
            Frag16 b00, b01;
            b00.q[0] = *(const uint4*)(kr0);
            b00.q[1] = *(const uint4*)(kr0 + 8);
            b01.q[0] = *(const uint4*)(kr0 + 32);
            b01.q[1] = *(const uint4*)(kr0 + 40);
            s0.v = __builtin_amdgcn_wmma_f32_16x16x32_bf16(
                false, aq0.v, false, b00.v, (short)0, s0.v, false, false);
            s0.v = __builtin_amdgcn_wmma_f32_16x16x32_bf16(
                false, aq1.v, false, b01.v, (short)0, s0.v, false, false);
            const ushort_t* kr1 = kr0 + 16 * 64;
            Frag16 b10, b11;
            b10.q[0] = *(const uint4*)(kr1);
            b10.q[1] = *(const uint4*)(kr1 + 8);
            b11.q[0] = *(const uint4*)(kr1 + 32);
            b11.q[1] = *(const uint4*)(kr1 + 40);
            s1.v = __builtin_amdgcn_wmma_f32_16x16x32_bf16(
                false, aq0.v, false, b10.v, (short)0, s1.v, false, false);
            s1.v = __builtin_amdgcn_wmma_f32_16x16x32_bf16(
                false, aq1.v, false, b11.v, (short)0, s1.v, false, false);
        }

        // --- masked online softmax; C layout: lane = key col, VGPR r = row ---
        const int mk0 = mb[kb + lr];
        const int mk1 = mb[kb + 16 + lr];
        float cscale[8];
#pragma unroll
        for (int r = 0; r < 8; ++r) {
            float v0 = mk0 ? s0.f[r] * scale : -1e30f;
            float v1 = mk1 ? s1.f[r] * scale : -1e30f;
            float mx = fmaxf(v0, v1);
            mx = fmaxf(mx, __shfl_xor(mx, 1));
            mx = fmaxf(mx, __shfl_xor(mx, 2));
            mx = fmaxf(mx, __shfl_xor(mx, 4));
            mx = fmaxf(mx, __shfl_xor(mx, 8));  // 16-lane group == one row set
            float nm = fmaxf(run_max[r], mx);
            float corr = __expf(run_max[r] - nm);
            run_max[r] = nm;
            float p0 = mk0 ? __expf(v0 - nm) : 0.0f;
            float p1 = mk1 ? __expf(v1 - nm) : 0.0f;
            s0.f[r] = p0; s1.f[r] = p1;
            float bs = p0 + p1;
            bs += __shfl_xor(bs, 1);
            bs += __shfl_xor(bs, 2);
            bs += __shfl_xor(bs, 4);
            bs += __shfl_xor(bs, 8);
            run_sum[r] = run_sum[r] * corr + bs;
            cscale[r] = corr;
        }
#pragma unroll
        for (int j = 0; j < 4; ++j)
#pragma unroll
            for (int r = 0; r < 8; ++r) acc[j].f[r] *= cscale[r];

        // --- P: C-frag -> LDS -> A-frag re-layout (wave-private region) ---
#pragma unroll
        for (int r = 0; r < 8; ++r) {
            const int mrow = r + (lh << 3);
            pl[mrow * 32 + lr] = f2bf(s0.f[r]);
            pl[mrow * 32 + 16 + lr] = f2bf(s1.f[r]);
        }
        Frag16 pa;
        {
            const ushort_t* prow = pl + lr * 32 + lh * 16;
            pa.q[0] = *(const uint4*)(prow);
            pa.q[1] = *(const uint4*)(prow + 8);
        }

        // --- acc += P * V  (V tile rows are contiguous in key dim) ---
#pragma unroll
        for (int j = 0; j < 4; ++j) {
            const ushort_t* vrow = vt + (j * 16 + lr) * 32 + lh * 16;
            Frag16 bv;
            bv.q[0] = *(const uint4*)(vrow);
            bv.q[1] = *(const uint4*)(vrow + 8);
            acc[j].v = __builtin_amdgcn_wmma_f32_16x16x32_bf16(
                false, pa.v, false, bv.v, (short)0, acc[j].v, false, false);
        }
    }

    // --- normalize and store y (bf16, [B,T,C]) ---
#pragma unroll
    for (int r = 0; r < 8; ++r) run_sum[r] = 1.0f / run_sum[r];
    const int t0 = qt * 16;
#pragma unroll
    for (int j = 0; j < 4; ++j)
#pragma unroll
        for (int r = 0; r < 8; ++r) {
            const int t = t0 + r + (lh << 3);
            const int c = h * HDd + j * 16 + lr;
            Y[((size_t)b * Tt + t) * Cc + c] = f2bf(acc[j].f[r] * run_sum[r]);
        }
}

// ---------------------------------------------------------------------------
extern "C" void kernel_launch(void* const* d_in, const int* in_sizes, int n_in,
                              void* d_out, int out_size, void* d_ws,
                              size_t ws_size, hipStream_t stream) {
    (void)in_sizes; (void)n_in; (void)out_size; (void)ws_size;
    const float* x    = (const float*)d_in[0];
    const int*   mask = (const int*)d_in[1];
    const float* Wq = (const float*)d_in[2];
    const float* bq = (const float*)d_in[3];
    const float* Wk = (const float*)d_in[4];
    const float* bk = (const float*)d_in[5];
    const float* Wv = (const float*)d_in[6];
    const float* bv = (const float*)d_in[7];
    const float* Wp = (const float*)d_in[8];
    const float* bp = (const float*)d_in[9];

    constexpr int XN = GM * GK;             // 3145728
    constexpr int WN = GN * GK;             // 589824
    constexpr int QN = Bb * Hh * Tt * HDd;  // 3145728

    ushort_t* ws   = (ushort_t*)d_ws;
    ushort_t* xbf  = ws;
    ushort_t* wqbf = xbf + XN;
    ushort_t* wkbf = wqbf + WN;
    ushort_t* wvbf = wkbf + WN;
    ushort_t* wpbf = wvbf + WN;
    ushort_t* qws  = wpbf + WN;
    ushort_t* kws  = qws + QN;
    ushort_t* vtws = kws + QN;
    ushort_t* yws  = vtws + QN;

    // Phase 0: convert activations + weights to bf16
    f32_to_bf16_kernel<<<(XN + 255) / 256, 256, 0, stream>>>(x, xbf, XN);
    f32_to_bf16_kernel<<<(WN + 255) / 256, 256, 0, stream>>>(Wq, wqbf, WN);
    f32_to_bf16_kernel<<<(WN + 255) / 256, 256, 0, stream>>>(Wk, wkbf, WN);
    f32_to_bf16_kernel<<<(WN + 255) / 256, 256, 0, stream>>>(Wv, wvbf, WN);
    f32_to_bf16_kernel<<<(WN + 255) / 256, 256, 0, stream>>>(Wp, wpbf, WN);

    dim3 ggrid(GM / 256, GN / 64);  // 16 x 12
    // Phase 1: Q, K -> [B,H,T,HD]; V -> [B,H,HD,T]
    gemm_wmma_kernel<MODE_QK><<<ggrid, 256, 0, stream>>>(xbf, wqbf, bq, qws);
    gemm_wmma_kernel<MODE_QK><<<ggrid, 256, 0, stream>>>(xbf, wkbf, bk, kws);
    gemm_wmma_kernel<MODE_VT><<<ggrid, 256, 0, stream>>>(xbf, wvbf, bv, vtws);

    // Phase 2: flash attention. B*H*(T/16)=3072 wave-tiles, 4 waves/block
    attn_kernel<<<(Bb * Hh * (Tt / 16)) / 4, 128, 0, stream>>>(qws, kws, vtws,
                                                               mask, yws);

    // Phase 3: output projection -> f32 d_out
    gemm_wmma_kernel<MODE_F32><<<ggrid, 256, 0, stream>>>(yws, wpbf, bp,
                                                          (float*)d_out);
}
